// GeometricAtomDecoder_44487271252535
// MI455X (gfx1250) — matrix operand order repaired and checked by reading, not compiled, not run on hardware
//
#include <hip/hip_runtime.h>
#include <hip/hip_bf16.h>
#include <math.h>

// ---------------- problem constants (from reference) ----------------
constexpr int B_   = 2;
constexpr int L_   = 512;
constexpr int C_   = 256;     // C_ATOM
constexpr int NA_  = 2048;    // 4*L atoms per batch
constexpr int H_   = 8;
constexpr int HD_  = 32;
constexpr int F_   = 1024;
constexpr int NL_  = 6;
constexpr int MROWS = B_ * NA_;  // 4096 flattened token rows

typedef _Float16 v16h __attribute__((ext_vector_type(16)));
typedef _Float16 v8h  __attribute__((ext_vector_type(8)));
typedef _Float16 v2h  __attribute__((ext_vector_type(2)));
typedef float    v8f  __attribute__((ext_vector_type(8)));

__device__ const float BBC[12] = {-1.458f, 0.f, 0.f,
                                   0.f,    0.f, 0.f,
                                   1.525f, 0.f, 0.f,
                                   2.4f,   1.f, 0.f};

__device__ inline float gelu_f(float x) {
  return 0.5f * x * (1.f + erff(x * 0.70710678118654752f));
}

// ---------------- CDNA5 async memory -> LDS copy (ASYNCcnt-tracked) --------
// Each lane copies 16B from its global address to its LDS offset; the copy
// engine runs concurrently with VALU/WMMA work. Generic pointers to __shared__
// carry the wave-relative LDS offset in their low 32 bits (LDS aperture base
// has zero low bits), which is exactly the VDST operand the instruction wants.
__device__ inline void async_ld_b128(void* lds_ptr, const _Float16* gptr) {
  unsigned lds_off = (unsigned)(unsigned long long)lds_ptr;
  asm volatile("global_load_async_to_lds_b128 %0, %1, off"
               :: "v"(lds_off), "v"(gptr)
               : "memory");
}
__device__ inline void wait_async() {
  asm volatile("s_wait_asynccnt 0x0" ::: "memory");
}

// ---------------- DPP16 butterfly reductions over 16 lanes ------------------
template <int CTRL>
__device__ inline float dpp_mov(float v) {
  return __int_as_float(__builtin_amdgcn_update_dpp(
      0, __float_as_int(v), CTRL, 0xf, 0xf, true));
}
__device__ inline float red_max16(float v) {
  v = fmaxf(v, dpp_mov<0xB1>(v));    // quad xor1
  v = fmaxf(v, dpp_mov<0x4E>(v));    // quad xor2
  v = fmaxf(v, dpp_mov<0x141>(v));   // row_half_mirror
  v = fmaxf(v, dpp_mov<0x140>(v));   // row_mirror
  return v;
}
__device__ inline float red_sum16(float v) {
  v += dpp_mov<0xB1>(v);
  v += dpp_mov<0x4E>(v);
  v += dpp_mov<0x141>(v);
  v += dpp_mov<0x140>(v);
  return v;
}

// ---------------- WMMA fragment helpers (ISA 7.12.2 layouts) ----------------
__device__ inline v16h frag_a(const _Float16* p, int ld) {
  int lane = threadIdx.x & 31;
  int m = lane & 15, hfv = lane >> 4;
  v16h f;
#pragma unroll
  for (int v = 0; v < 8; ++v) {
    int kb = ((v & 4) << 2) + hfv * 8 + (v & 3) * 2;
    v2h t = *(const v2h*)&p[m * ld + kb];
    f[2 * v]     = t[0];
    f[2 * v + 1] = t[1];
  }
  return f;
}
__device__ inline v16h frag_b(const _Float16* p, int ld) {
  int lane = threadIdx.x & 31;
  int n = lane & 15, kb = (lane >> 4) * 8;
  v8h lo = *(const v8h*)&p[n * ld + kb];
  v8h hi = *(const v8h*)&p[n * ld + kb + 16];
  return __builtin_shufflevector(lo, hi, 0, 1, 2, 3, 4, 5, 6, 7,
                                 8, 9, 10, 11, 12, 13, 14, 15);
}
__device__ inline v8f wmma16(v16h a, v16h b, v8f c) {
  return __builtin_amdgcn_wmma_f32_16x16x32_f16(false, a, false, b, (short)0, c,
                                                false, false);
}

// ---------------- f32 -> f16 weight conversion ----------------
__global__ void cvt_f16_kernel(const float* __restrict__ s, _Float16* __restrict__ d, int n) {
  int i = blockIdx.x * 256 + threadIdx.x;
  if (i < n) d[i] = (_Float16)s[i];
}

// ---------------- pack residue mask (bool/byte) into bit words --------------
__global__ void mask_bits_kernel(const unsigned char* __restrict__ m,
                                 unsigned* __restrict__ bits, int nres) {
  int w = blockIdx.x * 32 + threadIdx.x;
  if (w * 32 < nres) {
    unsigned v = 0;
#pragma unroll
    for (int i = 0; i < 32; ++i) v |= (m[w * 32 + i] ? 1u : 0u) << i;
    bits[w] = v;
  }
}

// ---------------- embedding: res_to_atom + atom_type + pos MLP --------------
__global__ void __launch_bounds__(256) embed_kernel(
    const float* __restrict__ trunk, const float* __restrict__ anchor,
    const float* __restrict__ rw, const float* __restrict__ rb,
    const float* __restrict__ atype, const float* __restrict__ pw1,
    const float* __restrict__ pb1, const float* __restrict__ pw2,
    const float* __restrict__ pb2, const float* __restrict__ unk,
    float* __restrict__ x) {
  int bl = blockIdx.x;      // b*L + l
  int t = threadIdx.x;
  __shared__ float tok[256];
  __shared__ float p1s[256];
  tok[t] = trunk[(size_t)bl * 256 + t];
  float ax = anchor[bl * 3 + 0], ay = anchor[bl * 3 + 1], az = anchor[bl * 3 + 2];
  bool anch = (fabsf(ax) + fabsf(ay) + fabsf(az)) > 1e-6f;
  float h1 = pw1[t * 3 + 0] * ax + pw1[t * 3 + 1] * ay + pw1[t * 3 + 2] * az + pb1[t];
  p1s[t] = gelu_f(h1);
  __syncthreads();
  float res = rb[t];
#pragma unroll 4
  for (int k = 0; k < 256; ++k) res += rw[t * 256 + k] * tok[k];
  float pf;
  if (anch) {
    pf = pb2[t];
#pragma unroll 4
    for (int k = 0; k < 256; ++k) pf += pw2[t * 256 + k] * p1s[k];
  } else {
    pf = unk[t];
  }
#pragma unroll
  for (int a = 0; a < 4; ++a)
    x[((size_t)bl * 4 + a) * 256 + t] = res + atype[a * 256 + t] + pf;
}

// ---------------- LayerNorm row -> f16 ----------------
__global__ void __launch_bounds__(256) ln_f16_kernel(
    const float* __restrict__ x, const float* __restrict__ w,
    const float* __restrict__ b, _Float16* __restrict__ out) {
  int row = blockIdx.x, t = threadIdx.x;
  __shared__ float red[256];
  float v = x[(size_t)row * 256 + t];
  red[t] = v;
  __syncthreads();
  for (int s2 = 128; s2 > 0; s2 >>= 1) { if (t < s2) red[t] += red[t + s2]; __syncthreads(); }
  float mean = red[0] * (1.f / 256.f);
  __syncthreads();
  float d = v - mean;
  red[t] = d * d;
  __syncthreads();
  for (int s2 = 128; s2 > 0; s2 >>= 1) { if (t < s2) red[t] += red[t + s2]; __syncthreads(); }
  float var = red[0] * (1.f / 256.f);
  float nv = d * rsqrtf(var + 1e-5f) * w[t] + b[t];
  out[(size_t)row * 256 + t] = (_Float16)nv;
}

// ---------------- generic WMMA GEMM:  out[m,n] = act[m,:] . W[n,:] + bias[n] -----
// Block tile 64(M) x 256(N); 8 waves, wave tile 32x64 -> 8 WMMAs per barrier
// pair. LDS double buffering with async memory->LDS copies: the copy engine
// stages tile t+1 while the WMMAs consume tile t.
// OP=0: store f16   OP=1: gelu -> f16   OP=2: resid[m,n] += val (f32 residual add)
template <int OP>
__global__ void __launch_bounds__(256) gemm_wmma_kernel(
    const _Float16* __restrict__ A, const _Float16* __restrict__ W,
    const float* __restrict__ bias, _Float16* __restrict__ out16,
    float* __restrict__ resid, int M, int N, int K) {
  __shared__ _Float16 As[2][64][40];    // row stride 80B (multiple of 16B)
  __shared__ _Float16 Bs[2][256][40];
  int m0 = blockIdx.x * 64, n0 = blockIdx.y * 256;
  int t = threadIdx.x, lane = t & 31, wave = t >> 5;
  int wm = wave >> 2, wn = wave & 3;   // wave tile 32(M) x 64(N)
  int arow = t >> 2, acol = (t & 3) * 8;
  v8f acc[2][4] = {};

  const _Float16* aP = &A[(size_t)(m0 + arow) * K + acol];
  const _Float16* bP = &W[(size_t)(n0 + t) * K];   // one B row per thread

  auto stage = [&](int k0, int buf) {
    async_ld_b128(&As[buf][arow][acol], aP + k0);
    async_ld_b128(&Bs[buf][t][0],  bP + k0);
    async_ld_b128(&Bs[buf][t][8],  bP + k0 + 8);
    async_ld_b128(&Bs[buf][t][16], bP + k0 + 16);
    async_ld_b128(&Bs[buf][t][24], bP + k0 + 24);
  };

  stage(0, 0);
  wait_async();
  __syncthreads();

  for (int k0 = 0; k0 < K; k0 += 32) {
    int cur = (k0 >> 5) & 1;
    if (k0 + 32 < K) stage(k0 + 32, cur ^ 1);   // async copy of next tile
    v16h a0 = frag_a(&As[cur][wm * 32][0], 40);
    v16h a1 = frag_a(&As[cur][wm * 32 + 16][0], 40);
#pragma unroll
    for (int ni = 0; ni < 4; ++ni) {
      v16h bfr = frag_b(&Bs[cur][wn * 64 + ni * 16][0], 40);
      acc[0][ni] = wmma16(a0, bfr, acc[0][ni]);
      acc[1][ni] = wmma16(a1, bfr, acc[1][ni]);
    }
    wait_async();       // next tile landed in LDS
    __syncthreads();    // all waves done reading current tile
  }

  int cn = lane & 15, hf = lane >> 4;
#pragma unroll
  for (int mi = 0; mi < 2; ++mi)
#pragma unroll
    for (int ni = 0; ni < 4; ++ni) {
      int col = n0 + wn * 64 + ni * 16 + cn;
      float bv = bias[col];
#pragma unroll
      for (int r = 0; r < 8; ++r) {
        int m = m0 + wm * 32 + mi * 16 + r + hf * 8;
        float v = acc[mi][ni][r] + bv;
        if (OP == 1) v = gelu_f(v);
        if (OP == 2) resid[(size_t)m * N + col] += v;
        else out16[(size_t)m * N + col] = (_Float16)v;
      }
    }
}

// ---------------- flash attention ------------------------------------------
// grid (NA/128, H, B), block 128 (4 waves, one wave per 32 query rows).
// K chunks staged by async memory->LDS copies (double-buffered, overlapping
// the softmax of the current chunk); V is staged through VGPRs because it is
// stored transposed in LDS so the P.V B-fragments read contiguously.
__global__ void __launch_bounds__(128) attn_kernel(
    const _Float16* __restrict__ Q, const _Float16* __restrict__ Kx,
    const _Float16* __restrict__ V, const float* __restrict__ pbias,
    const unsigned* __restrict__ mbits, _Float16* __restrict__ O, int layer) {
  int b = blockIdx.z, h = blockIdx.y;
  int wave = threadIdx.x >> 5, lane = threadIdx.x & 31;
  int qbase = blockIdx.x * 128 + wave * 32;
  int cn = lane & 15, hf = lane >> 4;
  __shared__ _Float16 P[4][32][40];      // per-wave 32x32 probability tile
  __shared__ _Float16 Ks[2][32][40];     // staged K chunk  [key][dim]
  __shared__ _Float16 Vt[2][32][40];     // staged V chunk, transposed [dim][key]

  const _Float16* qp = Q + ((size_t)(b * NA_ + qbase)) * C_ + h * HD_;
  v16h qa0 = frag_a(qp, C_);
  v16h qa1 = frag_a(qp + 16 * C_, C_);

  float pb[16];
#pragma unroll
  for (int e = 0; e < 16; ++e) pb[e] = pbias[((size_t)layer * H_ + h) * 16 + e];

  float rmax[2][8], rsum[2][8];
#pragma unroll
  for (int mi = 0; mi < 2; ++mi)
#pragma unroll
    for (int r = 0; r < 8; ++r) { rmax[mi][r] = -1e30f; rsum[mi][r] = 0.f; }
  v8f oacc[2][2] = {};
  const float scale = 0.17677669529663687f;   // 1/sqrt(32)

  int skey = threadIdx.x >> 2;          // 0..31 staging key row
  int sdc  = (threadIdx.x & 3) * 8;     // staging dim column
  const _Float16* kbase = Kx + ((size_t)b * NA_) * C_ + h * HD_;
  const _Float16* vbase = V  + ((size_t)b * NA_) * C_ + h * HD_;

  auto stage = [&](int kc, int buf) {
    async_ld_b128(&Ks[buf][skey][sdc], &kbase[(size_t)(kc + skey) * C_ + sdc]);
    v8h vv = *(const v8h*)&vbase[(size_t)(kc + skey) * C_ + sdc];
#pragma unroll
    for (int e = 0; e < 8; ++e) Vt[buf][sdc + e][skey] = vv[e];   // transpose
  };

  stage(0, 0);
  wait_async();
  __syncthreads();

  for (int kc = 0; kc < NA_; kc += 32) {
    int cur = (kc >> 5) & 1;
    v16h kb0 = frag_b(&Ks[cur][0][0], 40);
    v16h kb1 = frag_b(&Ks[cur][16][0], 40);
    v8f z = {0.f, 0.f, 0.f, 0.f, 0.f, 0.f, 0.f, 0.f};
    v8f sAcc[2][2];
    sAcc[0][0] = wmma16(qa0, kb0, z);
    sAcc[0][1] = wmma16(qa0, kb1, z);
    sAcc[1][0] = wmma16(qa1, kb0, z);
    sAcc[1][1] = wmma16(qa1, kb1, z);

    if (kc + 32 < NA_) stage(kc + 32, cur ^ 1);   // overlap with softmax

    // residue mask: keys kc..kc+31 span 8 bits inside one 32-bit word
    int rbase = b * L_ + (kc >> 2);
    unsigned mw = mbits[rbase >> 5];         // wave-uniform -> scalar load
    int bofs = rbase & 31;
    bool mk0 = (mw >> (bofs + (cn >> 2))) & 1u;
    bool mk1 = (mw >> (bofs + 4 + (cn >> 2))) & 1u;

#pragma unroll
    for (int mi = 0; mi < 2; ++mi) {
      float v0[8], v1[8];
#pragma unroll
      for (int r = 0; r < 8; ++r) {
        float bb = pb[(r & 3) * 4 + (cn & 3)];   // bias[i%4, j%4]
        v0[r] = mk0 ? sAcc[mi][0][r] * scale + bb : -1e30f;
        v1[r] = mk1 ? sAcc[mi][1][r] * scale + bb : -1e30f;
      }
#pragma unroll
      for (int r = 0; r < 8; ++r) {
        float mx = red_max16(fmaxf(v0[r], v1[r]));
        float nm = fmaxf(rmax[mi][r], mx);
        float alpha = (rmax[mi][r] < -1e29f) ? 0.f : __expf(rmax[mi][r] - nm);
        rmax[mi][r] = nm;
        float p0, p1;
        if (nm < -1e29f) { p0 = 0.f; p1 = 0.f; }
        else { p0 = __expf(v0[r] - nm); p1 = __expf(v1[r] - nm); }
        float ps = red_sum16(p0 + p1);
        rsum[mi][r] = rsum[mi][r] * alpha + ps;
        oacc[mi][0][r] *= alpha;
        oacc[mi][1][r] *= alpha;
        P[wave][mi * 16 + r + hf * 8][cn]      = (_Float16)p0;
        P[wave][mi * 16 + r + hf * 8][16 + cn] = (_Float16)p1;
      }
    }
    __syncthreads();   // P written; Vt staging stores visible
    v16h pa0 = frag_a(&P[wave][0][0], 40);
    v16h pa1 = frag_a(&P[wave][16][0], 40);
    v16h vb0 = frag_b(&Vt[cur][0][0], 40);    // dims 0..15
    v16h vb1 = frag_b(&Vt[cur][16][0], 40);   // dims 16..31
    oacc[0][0] = wmma16(pa0, vb0, oacc[0][0]);
    oacc[0][1] = wmma16(pa0, vb1, oacc[0][1]);
    oacc[1][0] = wmma16(pa1, vb0, oacc[1][0]);
    oacc[1][1] = wmma16(pa1, vb1, oacc[1][1]);
    wait_async();      // async K copy for next chunk completed
    __syncthreads();   // P/Vt reads done before next overwrite
  }

#pragma unroll
  for (int mi = 0; mi < 2; ++mi)
#pragma unroll
    for (int r = 0; r < 8; ++r) {
      float den = rsum[mi][r];
      float inv = den > 0.f ? 1.f / den : 0.f;
      int m = qbase + mi * 16 + r + hf * 8;
      size_t base = ((size_t)(b * NA_ + m)) * C_ + h * HD_;
      O[base + cn]      = (_Float16)(oacc[mi][0][r] * inv);
      O[base + 16 + cn] = (_Float16)(oacc[mi][1][r] * inv);
    }
}

// ---------------- final: output LN + 3-dim projection + prior coords -------
__global__ void __launch_bounds__(256) final_kernel(
    const float* __restrict__ x, const float* __restrict__ nw,
    const float* __restrict__ nb, const float* __restrict__ pw,
    const float* __restrict__ pbv, const float* __restrict__ anchor,
    const float* __restrict__ bboff, float* __restrict__ out) {
  int row = blockIdx.x;           // 0 .. B*NA-1
  int b = row / NA_, n = row % NA_, l = n >> 2, a = n & 3;
  int t = threadIdx.x;
  __shared__ float red[256];
  __shared__ float res3[3];
  float v = x[(size_t)row * 256 + t];
  red[t] = v;
  __syncthreads();
  for (int s2 = 128; s2 > 0; s2 >>= 1) { if (t < s2) red[t] += red[t + s2]; __syncthreads(); }
  float mean = red[0] * (1.f / 256.f);
  __syncthreads();
  float d = v - mean;
  red[t] = d * d;
  __syncthreads();
  for (int s2 = 128; s2 > 0; s2 >>= 1) { if (t < s2) red[t] += red[t + s2]; __syncthreads(); }
  float var = red[0] * (1.f / 256.f);
  float nv = d * rsqrtf(var + 1e-5f) * nw[t] + nb[t];
  for (int dd = 0; dd < 3; ++dd) {
    __syncthreads();
    red[t] = nv * pw[dd * 256 + t];
    __syncthreads();
    for (int s2 = 128; s2 > 0; s2 >>= 1) { if (t < s2) red[t] += red[t + s2]; __syncthreads(); }
    if (t == 0) res3[dd] = red[0] + pbv[dd];
  }
  __syncthreads();
  if (t < 3) {
    float prior = anchor[((size_t)b * L_ + l) * 3 + t] +
                  (BBC[a * 3 + t] + bboff[a * 3 + t]) * 0.1f;
    out[(((size_t)b * L_ + l) * 4 + a) * 3 + t] = prior + res3[t];
  }
}

// ---------------------------------------------------------------------------
extern "C" void kernel_launch(void* const* d_in, const int* in_sizes, int n_in,
                              void* d_out, int out_size, void* d_ws, size_t ws_size,
                              hipStream_t stream) {
  (void)in_sizes; (void)n_in; (void)out_size; (void)ws_size;
  const float* trunk  = (const float*)d_in[0];
  const float* anchor = (const float*)d_in[1];
  const unsigned char* mask = (const unsigned char*)d_in[2];
  const float* res_w = (const float*)d_in[3];
  const float* res_b = (const float*)d_in[4];
  const float* atype = (const float*)d_in[5];
  const float* pw1   = (const float*)d_in[6];
  const float* pb1   = (const float*)d_in[7];
  const float* pw2   = (const float*)d_in[8];
  const float* pb2   = (const float*)d_in[9];
  const float* unk   = (const float*)d_in[10];
  const float* bboff = (const float*)d_in[11];
  const float* onw   = (const float*)d_in[12];
  const float* onb   = (const float*)d_in[13];
  const float* opw   = (const float*)d_in[14];
  const float* opb   = (const float*)d_in[15];
  const float* ln_w  = (const float*)d_in[16];
  const float* ln_b  = (const float*)d_in[17];
  const float* q_w   = (const float*)d_in[18];
  const float* q_b   = (const float*)d_in[19];
  const float* k_w   = (const float*)d_in[20];
  const float* k_b   = (const float*)d_in[21];
  const float* v_w   = (const float*)d_in[22];
  const float* v_b   = (const float*)d_in[23];
  const float* o_w   = (const float*)d_in[24];
  const float* o_b   = (const float*)d_in[25];
  const float* pbias = (const float*)d_in[26];
  const float* fln_w = (const float*)d_in[27];
  const float* fln_b = (const float*)d_in[28];
  const float* f1_w  = (const float*)d_in[29];
  const float* f1_b  = (const float*)d_in[30];
  const float* f2_w  = (const float*)d_in[31];
  const float* f2_b  = (const float*)d_in[32];

  char* wsp = (char*)d_ws;
  auto carve = [&](size_t bytes) -> void* {
    void* p = (void*)wsp;
    wsp += (bytes + 255) & ~(size_t)255;
    return p;
  };
  float*    x     = (float*)carve((size_t)MROWS * C_ * 4);
  _Float16* h16   = (_Float16*)carve((size_t)MROWS * C_ * 2);
  _Float16* q16   = (_Float16*)carve((size_t)MROWS * C_ * 2);
  _Float16* k16   = (_Float16*)carve((size_t)MROWS * C_ * 2);
  _Float16* v16b  = (_Float16*)carve((size_t)MROWS * C_ * 2);
  _Float16* ao16  = (_Float16*)carve((size_t)MROWS * C_ * 2);
  _Float16* ff16  = (_Float16*)carve((size_t)MROWS * F_ * 2);
  _Float16* wq16  = (_Float16*)carve((size_t)NL_ * C_ * C_ * 2);
  _Float16* wk16  = (_Float16*)carve((size_t)NL_ * C_ * C_ * 2);
  _Float16* wv16  = (_Float16*)carve((size_t)NL_ * C_ * C_ * 2);
  _Float16* wo16  = (_Float16*)carve((size_t)NL_ * C_ * C_ * 2);
  _Float16* wf1   = (_Float16*)carve((size_t)NL_ * F_ * C_ * 2);
  _Float16* wf2   = (_Float16*)carve((size_t)NL_ * C_ * F_ * 2);
  unsigned* mbits = (unsigned*)carve((size_t)(B_ * L_ / 32) * 4);

  auto cvt = [&](const float* s, _Float16* dptr, int n) {
    cvt_f16_kernel<<<(n + 255) / 256, 256, 0, stream>>>(s, dptr, n);
  };
  cvt(q_w, wq16, NL_ * C_ * C_);
  cvt(k_w, wk16, NL_ * C_ * C_);
  cvt(v_w, wv16, NL_ * C_ * C_);
  cvt(o_w, wo16, NL_ * C_ * C_);
  cvt(f1_w, wf1, NL_ * F_ * C_);
  cvt(f2_w, wf2, NL_ * C_ * F_);
  mask_bits_kernel<<<1, 32, 0, stream>>>(mask, mbits, B_ * L_);

  embed_kernel<<<B_ * L_, 256, 0, stream>>>(trunk, anchor, res_w, res_b, atype,
                                            pw1, pb1, pw2, pb2, unk, x);

  dim3 gC(MROWS / 64, C_ / 256);    // (64, 1)
  dim3 gF(MROWS / 64, F_ / 256);    // (64, 4)
  dim3 gAttn(NA_ / 128, H_, B_);
  for (int i = 0; i < NL_; ++i) {
    ln_f16_kernel<<<MROWS, 256, 0, stream>>>(x, ln_w + i * C_, ln_b + i * C_, h16);
    gemm_wmma_kernel<0><<<gC, 256, 0, stream>>>(h16, wq16 + (size_t)i * C_ * C_,
                                                q_b + i * C_, q16, nullptr, MROWS, C_, C_);
    gemm_wmma_kernel<0><<<gC, 256, 0, stream>>>(h16, wk16 + (size_t)i * C_ * C_,
                                                k_b + i * C_, k16, nullptr, MROWS, C_, C_);
    gemm_wmma_kernel<0><<<gC, 256, 0, stream>>>(h16, wv16 + (size_t)i * C_ * C_,
                                                v_b + i * C_, v16b, nullptr, MROWS, C_, C_);
    attn_kernel<<<gAttn, 128, 0, stream>>>(q16, k16, v16b, pbias, mbits, ao16, i);
    gemm_wmma_kernel<2><<<gC, 256, 0, stream>>>(ao16, wo16 + (size_t)i * C_ * C_,
                                                o_b + i * C_, nullptr, x, MROWS, C_, C_);
    ln_f16_kernel<<<MROWS, 256, 0, stream>>>(x, fln_w + i * C_, fln_b + i * C_, h16);
    gemm_wmma_kernel<1><<<gF, 256, 0, stream>>>(h16, wf1 + (size_t)i * F_ * C_,
                                                f1_b + i * F_, ff16, nullptr, MROWS, F_, C_);
    gemm_wmma_kernel<2><<<gC, 256, 0, stream>>>(ff16, wf2 + (size_t)i * C_ * F_,
                                                f2_b + i * C_, nullptr, x, MROWS, C_, F_);
  }
  final_kernel<<<MROWS, 256, 0, stream>>>(x, onw, onb, opw, opb, anchor, bboff,
                                          (float*)d_out);
}
